// Grid2MeshEdgeUpdate_21998822490253
// MI455X (gfx1250) — matrix operand order repaired and checked by reading, not compiled, not run on hardware
//
#include <hip/hip_runtime.h>

typedef unsigned short ushort_t;
typedef __attribute__((ext_vector_type(16))) __bf16 v16bf;
typedef __attribute__((ext_vector_type(8)))  float  v8f;
typedef int v4i_t __attribute__((vector_size(16)));   // matches builtin param type

#define E_TOT   131072
#define TILE_M  64
#define NBLK    (E_TOT / TILE_M)   // 2048

// ---------- LDS layout (bytes) ----------
#define PA   784      // A row pitch   (392 bf16)  -> 196 dwords % 64 = 4 (conflict-free)
#define PH1  1040     // h1 row pitch  (520 bf16)  -> 260 dwords % 64 = 4
#define PB1  784      // staged W1 column pitch (K=384 + pad8)
#define PW2  1040     // staged W2 column pitch (K=512 + pad8)
#define PW3  144      // staged W3 column pitch (K=64  + pad8)
#define PH2  144      // h2 row pitch  (72 bf16)

#define A_OFF   0
#define A_BYTES (64 * PA)               // 50176
#define H1_OFF  (A_OFF + A_BYTES)
#define H1_BYTES (64 * PH1)             // 66560
#define B_OFF   (H1_OFF + H1_BYTES)     // 116736
#define B_BYTES (32 * PW2)              // 33280 = max(32*PB1, 32*PW2, 128*PW3)
#define H2_OFF  (B_OFF + B_BYTES)       // 150016
#define H2_BYTES (64 * PH2)             // 9216
#define SMEM_BYTES (H2_OFF + H2_BYTES)  // 159232  (<= 160KB)

// ---------- workspace layout (bytes) ----------
#define W1T_OFF  0                      // 196608 bf16 (W1^T: [512][384])
#define W2T_OFF  393216                 // 32768  bf16 (W2^T: [64][512])
#define W3T_OFF  458752                 // 8192   bf16 (W3^T: [128][64])
#define PSUM_OFF 475136                 // float[2048]
#define PSQ_OFF  483328                 // float[2048]
#define STAT_OFF 491520                 // float[2]  (mean, rstd)

// CDNA5 async copy to LDS (ASYNCcnt-tracked, bypasses VGPRs) when the
// toolchain declares the builtins; otherwise identical explicit copy.
#if defined(__HIP_DEVICE_COMPILE__) && \
    __has_builtin(__builtin_amdgcn_global_load_async_to_lds_b128) && \
    __has_builtin(__builtin_amdgcn_s_wait_asynccnt)
#define HAVE_ASYNC_LDS 1
#else
#define HAVE_ASYNC_LDS 0
#endif

__device__ __forceinline__ void stage_cp16(const ushort_t* gsrc, unsigned char* ldst) {
#if HAVE_ASYNC_LDS
  __builtin_amdgcn_global_load_async_to_lds_b128(
      (__attribute__((address_space(1))) v4i_t*)gsrc,
      (__attribute__((address_space(3))) v4i_t*)ldst, 0, 0);
#else
  *(uint4*)ldst = *(const uint4*)gsrc;
#endif
}
__device__ __forceinline__ void stage_wait() {
#if HAVE_ASYNC_LDS
  __builtin_amdgcn_s_wait_asynccnt(0);
#endif
}

__device__ __forceinline__ ushort_t f2bf(float f) {
  unsigned u = __builtin_bit_cast(unsigned, f);
  u += 0x7FFFu + ((u >> 16) & 1u);      // round-to-nearest-even
  return (ushort_t)(u >> 16);
}
__device__ __forceinline__ unsigned pk2(float a, float b) {
  return (unsigned)f2bf(a) | ((unsigned)f2bf(b) << 16);
}
__device__ __forceinline__ float silu(float x) {
  return x * (1.0f / (1.0f + __expf(-x)));
}

union FragU { uint4 u[2]; v16bf v; };
__device__ __forceinline__ v16bf ldfrag(const unsigned char* p, int off2) {
  FragU f;
  f.u[0] = *(const uint4*)p;
  f.u[1] = *(const uint4*)(p + off2);
  return f.v;
}
__device__ __forceinline__ v8f wmma_bf16(v16bf a, v16bf b, v8f c) {
  // (neg_a, A, neg_b, B, c_mod, C, reuse_a, reuse_b)
  return __builtin_amdgcn_wmma_f32_16x16x32_bf16(false, a, false, b, (short)0, c, false, false);
}

// ============================================================
// Kernel 0: convert + transpose weights to bf16 into workspace
// ============================================================
__global__ void __launch_bounds__(256)
convert_weights(const float* __restrict__ W1, const float* __restrict__ W2,
                const float* __restrict__ W3,
                ushort_t* __restrict__ w1t, ushort_t* __restrict__ w2t,
                ushort_t* __restrict__ w3t) {
  int i = blockIdx.x * 256 + threadIdx.x;
  if (i < 196608) {                         // W1^T [n=512][k=384]
    int n = i / 384, k = i - n * 384;
    w1t[i] = f2bf(W1[k * 512 + n]);
  } else if (i < 229376) {                  // W2^T [n=64][k=512]
    int j = i - 196608;
    int n = j / 512, k = j - n * 512;
    w2t[j] = f2bf(W2[k * 64 + n]);
  } else if (i < 237568) {                  // W3^T [n=128][k=64]
    int j = i - 229376;
    int n = j / 64, k = j - n * 64;
    w3t[j] = f2bf(W3[k * 128 + n]);
  }
}

// ============================================================
// Kernel 1: fused gather + 3-layer MLP (bf16 WMMA, f32 accum)
//           writes h3 to d_out, per-block sum/sumsq partials
// ============================================================
__global__ void __launch_bounds__(256)
fused_mlp(const float* __restrict__ gx, const float* __restrict__ mx,
          const int* __restrict__ g2i, const float* __restrict__ g2x,
          const float* __restrict__ b1p, const float* __restrict__ b2p,
          const float* __restrict__ b3p,
          const ushort_t* __restrict__ w1t, const ushort_t* __restrict__ w2t,
          const ushort_t* __restrict__ w3t,
          float* __restrict__ h3out, float* __restrict__ psum,
          float* __restrict__ psq) {
  __shared__ __align__(16) unsigned char smem[SMEM_BYTES];
  const int e0   = blockIdx.x * TILE_M;
  const int tid  = threadIdx.x;
  const int lane = tid & 31;
  const int wave = tid >> 5;
  const int lm   = lane & 15;      // row/col within 16-wide tile
  const int hi   = lane >> 4;      // half-wave
  const int rt   = wave & 3;       // row tile 0..3 (16 edges each)

  // ---- Stage A: gather gx[row] | mx[col] | g2me_x -> bf16 LDS rows of 384 ----
  for (int idx = tid; idx < TILE_M * 48; idx += 256) {
    int edge = idx / 48;
    int c    = idx - edge * 48;
    int seg  = c >> 4;             // 0:gx 1:mx 2:g2me_x
    int off  = (c & 15) * 8;       // 0..120, 8 floats per chunk
    const float* src;
    if (seg == 0)      src = gx  + (size_t)g2i[e0 + edge] * 128 + off;
    else if (seg == 1) src = mx  + (size_t)g2i[E_TOT + e0 + edge] * 128 + off;
    else               src = g2x + (size_t)(e0 + edge) * 128 + off;
    float4 f0 = *(const float4*)src;
    float4 f1 = *(const float4*)(src + 4);
    uint4 p;
    p.x = pk2(f0.x, f0.y); p.y = pk2(f0.z, f0.w);
    p.z = pk2(f1.x, f1.y); p.w = pk2(f1.z, f1.w);
    *(uint4*)(smem + A_OFF + edge * PA + (seg * 128 + off) * 2) = p;
  }
  __syncthreads();

  // ---- Cache this wave's 16x384 A-strip in registers (12 K-frags) ----
  v16bf areg[12];
  {
    const unsigned char* abase = smem + A_OFF + (rt * 16 + lm) * PA + hi * 16;
    #pragma unroll
    for (int kt = 0; kt < 12; ++kt) areg[kt] = ldfrag(abase + kt * 64, 32);
  }

  // ---- GEMM1: [64x384] @ [384x512] -> SiLU -> h1 bf16 LDS ----
  {
    const int ct1 = wave >> 2;                 // col tile within 32-col group
    for (int cg = 0; cg < 16; ++cg) {          // 16 groups of 32 columns
      __syncthreads();
      for (int idx = tid; idx < 32 * 48; idx += 256) {
        int j = idx / 48, q = idx - j * 48;
        stage_cp16(w1t + ((cg * 32 + j) * 384 + q * 8),
                   smem + B_OFF + j * PB1 + q * 16);
      }
      stage_wait();
      __syncthreads();
      v8f acc = {};
      #pragma unroll
      for (int kt = 0; kt < 12; ++kt) {
        v16bf b = ldfrag(smem + B_OFF + (ct1 * 16 + lm) * PB1 + kt * 64 + hi * 16, 32);
        acc = wmma_bf16(areg[kt], b, acc);
      }
      int col = cg * 32 + ct1 * 16 + lm;
      float bv = b1p[col];
      #pragma unroll
      for (int r = 0; r < 8; ++r) {
        int mrow = rt * 16 + r + hi * 8;
        *(ushort_t*)(smem + H1_OFF + mrow * PH1 + col * 2) = f2bf(silu(acc[r] + bv));
      }
    }
  }

  // ---- GEMM2: [64x512] @ [512x64] -> SiLU -> h2 bf16 LDS ----
  {
    const int ct2 = wave >> 2;                 // col tile within 32-col chunk
    for (int ch = 0; ch < 2; ++ch) {
      __syncthreads();
      for (int idx = tid; idx < 32 * 64; idx += 256) {
        int j = idx >> 6, q = idx & 63;
        stage_cp16(w2t + ((ch * 32 + j) * 512 + q * 8),
                   smem + B_OFF + j * PW2 + q * 16);
      }
      stage_wait();
      __syncthreads();
      v8f acc = {};
      #pragma unroll
      for (int kt = 0; kt < 16; ++kt) {
        v16bf a = ldfrag(smem + H1_OFF + (rt * 16 + lm) * PH1 + kt * 64 + hi * 16, 32);
        v16bf b = ldfrag(smem + B_OFF + (ct2 * 16 + lm) * PW2 + kt * 64 + hi * 16, 32);
        acc = wmma_bf16(a, b, acc);
      }
      int col = ch * 32 + ct2 * 16 + lm;
      float bv = b2p[col];
      #pragma unroll
      for (int r = 0; r < 8; ++r) {
        int mrow = rt * 16 + r + hi * 8;
        *(ushort_t*)(smem + H2_OFF + mrow * PH2 + col * 2) = f2bf(silu(acc[r] + bv));
      }
    }
  }

  // ---- GEMM3: [64x64] @ [64x128] + b3 -> h3 (f32) to global ----
  float lsum = 0.f, lsq = 0.f;
  {
    __syncthreads();
    for (int idx = tid; idx < 128 * 8; idx += 256) {
      int j = idx >> 3, q = idx & 7;
      stage_cp16(w3t + (j * 64 + q * 8),
                 smem + B_OFF + j * PW3 + q * 16);
    }
    stage_wait();
    __syncthreads();
    const int ct3 = (wave >> 2) * 4;           // 4 col tiles per wave
    v8f acc3[4] = {{}, {}, {}, {}};
    #pragma unroll
    for (int kt = 0; kt < 2; ++kt) {
      v16bf a = ldfrag(smem + H2_OFF + (rt * 16 + lm) * PH2 + kt * 64 + hi * 16, 32);
      #pragma unroll
      for (int j = 0; j < 4; ++j) {
        v16bf b = ldfrag(smem + B_OFF + ((ct3 + j) * 16 + lm) * PW3 + kt * 64 + hi * 16, 32);
        acc3[j] = wmma_bf16(a, b, acc3[j]);
      }
    }
    #pragma unroll
    for (int j = 0; j < 4; ++j) {
      int col = (ct3 + j) * 16 + lm;
      float bias = b3p[col];
      #pragma unroll
      for (int r = 0; r < 8; ++r) {
        int mrow = rt * 16 + r + hi * 8;
        float v = acc3[j][r] + bias;
        h3out[(size_t)(e0 + mrow) * 128 + col] = v;
        lsum += v;
        lsq  += v * v;
      }
    }
  }

  // ---- per-block reduction of sum / sumsq (A region reused as scratch) ----
  float* rs = (float*)(smem + A_OFF);
  float* rq = rs + 256;
  __syncthreads();
  rs[tid] = lsum; rq[tid] = lsq;
  __syncthreads();
  for (int s = 128; s > 0; s >>= 1) {
    if (tid < s) { rs[tid] += rs[tid + s]; rq[tid] += rq[tid + s]; }
    __syncthreads();
  }
  if (tid == 0) { psum[blockIdx.x] = rs[0]; psq[blockIdx.x] = rq[0]; }
}

// ============================================================
// Kernel 2: deterministic global mean / rstd (double accum)
// ============================================================
__global__ void __launch_bounds__(256)
stats_kernel(const float* __restrict__ psum, const float* __restrict__ psq,
             float* __restrict__ stats) {
  __shared__ double ss[256], sq[256];
  double a = 0.0, b = 0.0;
  for (int i = threadIdx.x; i < NBLK; i += 256) { a += psum[i]; b += psq[i]; }
  ss[threadIdx.x] = a; sq[threadIdx.x] = b;
  __syncthreads();
  for (int s = 128; s > 0; s >>= 1) {
    if (threadIdx.x < s) { ss[threadIdx.x] += ss[threadIdx.x + s];
                           sq[threadIdx.x] += sq[threadIdx.x + s]; }
    __syncthreads();
  }
  if (threadIdx.x == 0) {
    double N = (double)E_TOT * 128.0;
    double mean = ss[0] / N;
    double var  = sq[0] / N - mean * mean;
    stats[0] = (float)mean;
    stats[1] = (float)(1.0 / sqrt(var + 1e-5));
  }
}

// ============================================================
// Kernel 3: out = g2me_x + (h3 - mean) * rstd * ln_w + ln_b
// ============================================================
__global__ void __launch_bounds__(256)
finalize(const float* __restrict__ g2x, const float* __restrict__ lnw,
         const float* __restrict__ lnb, const float* __restrict__ stats,
         float* __restrict__ out, int n4) {
  int i = blockIdx.x * 256 + threadIdx.x;
  if (i >= n4) return;
  float mean = stats[0], rstd = stats[1];
  float4 h = ((const float4*)out)[i];
  float4 g = ((const float4*)g2x)[i];
  float4 w = ((const float4*)lnw)[i];
  float4 b = ((const float4*)lnb)[i];
  float4 o;
  o.x = g.x + (h.x - mean) * rstd * w.x + b.x;
  o.y = g.y + (h.y - mean) * rstd * w.y + b.y;
  o.z = g.z + (h.z - mean) * rstd * w.z + b.z;
  o.w = g.w + (h.w - mean) * rstd * w.w + b.w;
  ((float4*)out)[i] = o;
}

extern "C" void kernel_launch(void* const* d_in, const int* in_sizes, int n_in,
                              void* d_out, int out_size, void* d_ws, size_t ws_size,
                              hipStream_t stream) {
  const float* gx   = (const float*)d_in[0];
  const float* mx   = (const float*)d_in[1];
  const int*   g2i  = (const int*)d_in[4];
  const float* g2x  = (const float*)d_in[5];
  const float* W1   = (const float*)d_in[8];
  const float* b1   = (const float*)d_in[9];
  const float* W2   = (const float*)d_in[10];
  const float* b2   = (const float*)d_in[11];
  const float* W3   = (const float*)d_in[12];
  const float* b3   = (const float*)d_in[13];
  const float* lnw  = (const float*)d_in[14];
  const float* lnb  = (const float*)d_in[15];
  float* out = (float*)d_out;

  unsigned char* ws = (unsigned char*)d_ws;
  ushort_t* w1t = (ushort_t*)(ws + W1T_OFF);
  ushort_t* w2t = (ushort_t*)(ws + W2T_OFF);
  ushort_t* w3t = (ushort_t*)(ws + W3T_OFF);
  float* psum  = (float*)(ws + PSUM_OFF);
  float* psq   = (float*)(ws + PSQ_OFF);
  float* stats = (float*)(ws + STAT_OFF);

  convert_weights<<<928, 256, 0, stream>>>(W1, W2, W3, w1t, w2t, w3t);
  fused_mlp<<<NBLK, 256, 0, stream>>>(gx, mx, g2i, g2x, b1, b2, b3,
                                      w1t, w2t, w3t, out, psum, psq);
  stats_kernel<<<1, 256, 0, stream>>>(psum, psq, stats);
  int n4 = E_TOT * 128 / 4;
  finalize<<<n4 / 256, 256, 0, stream>>>(g2x, lnw, lnb, stats, out, n4);
}
